// BottleneckAdapter_76287209111716
// MI455X (gfx1250) — matrix operand order repaired and checked by reading, not compiled, not run on hardware
//
#include <hip/hip_runtime.h>
#include <hip/hip_bf16.h>
#include <math.h>

typedef __bf16 bf16;
typedef bf16          v16bf __attribute__((ext_vector_type(16)));
typedef float         v8f   __attribute__((ext_vector_type(8)));
typedef float         v4f   __attribute__((ext_vector_type(4)));
typedef unsigned int  v4u   __attribute__((ext_vector_type(4)));

union Frag {            // one WMMA A/B operand: 16 bf16 = 8 VGPRs
    v16bf v;
    bf16  e[16];
    v4u   q[2];
};

#define DIM   1024
#define BOT   64
#define MT    64        // tokens per block
#define HSTR  72        // padded LDS stride for H (bank-conflict-free)
#define EPS   1e-5f

__device__ __forceinline__ float gelu_exact(float x) {
    return 0.5f * x * (1.0f + erff(x * 0.70710678118654752f));
}

// ---------------------------------------------------------------------------
// Prep 1: wdT[n][k] = bf16( w_down[k][n] * norm_w[k] )   (n<64, k<1024)
//         wuT[n][k] = bf16( w_up[k][n] )                 (n<1024, k<64)
// ---------------------------------------------------------------------------
__global__ void prep_weights(const float* __restrict__ wd,
                             const float* __restrict__ wu,
                             const float* __restrict__ nw,
                             bf16* __restrict__ wdT,
                             bf16* __restrict__ wuT)
{
    int t = blockIdx.x * 256 + threadIdx.x;
    if (t < DIM * BOT) {
        int n = t >> 10, k = t & (DIM - 1);
        wdT[t] = (bf16)(wd[k * BOT + n] * nw[k]);
    } else {
        int u = t - DIM * BOT;
        int n = u >> 6, k = u & (BOT - 1);
        wuT[u] = (bf16)wu[(size_t)k * DIM + n];
    }
}

// ---------------------------------------------------------------------------
// Prep 2: bd_eff[n] = b_down[n] + sum_k norm_b[k] * w_down[k][n]
// ---------------------------------------------------------------------------
__global__ void prep_bias(const float* __restrict__ wd,
                          const float* __restrict__ nb,
                          const float* __restrict__ bd,
                          float* __restrict__ bd_eff)
{
    int n = threadIdx.x;           // 64 threads
    float s = bd[n];
    for (int k = 0; k < DIM; ++k) s += nb[k] * wd[k * BOT + n];
    bd_eff[n] = s;
}

// ---------------------------------------------------------------------------
// Fused LayerNorm -> down-proj (WMMA) -> GELU -> up-proj (WMMA) -> residual
// 256 threads = 8 waves, 64 tokens per block.
// ---------------------------------------------------------------------------
__global__ void __launch_bounds__(256)
adapter_kernel(const float* __restrict__ x,
               const bf16*  __restrict__ wdT,      // [64][1024] bf16, gamma-folded
               const bf16*  __restrict__ wuT,      // [1024][64] bf16
               const float* __restrict__ bd_eff,   // [64]
               const float* __restrict__ bup,      // [1024]
               const float* __restrict__ scale_p,
               float* __restrict__ out)
{
    __shared__ float smean[MT];
    __shared__ float srstd[MT];
    __shared__ __align__(16) bf16 Hs[MT * HSTR];   // GELU output, 64 x 64 (+pad)

    const int tid  = threadIdx.x;
    const int wv   = tid >> 5;          // wave 0..7
    const int lane = tid & 31;
    const int h    = lane >> 4;         // half-wave
    const int r    = lane & 15;
    const int rowbase = blockIdx.x * MT;
    const float scale = *scale_p;

    // ---------------- Phase 1: LN statistics (one wave per row) -------------
    for (int rr = wv; rr < MT; rr += 8) {
        const float* xrow = x + (size_t)(rowbase + rr) * DIM;
        float s = 0.f, s2 = 0.f;
        #pragma unroll
        for (int it = 0; it < 8; ++it) {
            v4f v = *(const v4f*)(xrow + it * 128 + lane * 4);
            s  += v.x + v.y + v.z + v.w;
            s2 += v.x * v.x + v.y * v.y + v.z * v.z + v.w * v.w;
        }
        #pragma unroll
        for (int m = 16; m >= 1; m >>= 1) {
            s  += __shfl_xor(s,  m, 32);
            s2 += __shfl_xor(s2, m, 32);
        }
        if (lane == 0) {
            float mu  = s * (1.0f / DIM);
            float var = s2 * (1.0f / DIM) - mu * mu;
            smean[rr] = mu;
            srstd[rr] = rsqrtf(var + EPS);
        }
    }
    __syncthreads();

    // ---------------- Phase 2: GEMM1  H = GELU(xhat @ wd + bd_eff) ----------
    // Output 64x64 -> 4 row-tiles x 4 col-tiles; wave wv owns row-tile wv>>1,
    // col-tiles {2*(wv&1), 2*(wv&1)+1}.
    const int tr = wv >> 1;             // row tile 0..3
    const int cb = (wv & 1) * 32;       // first of two 16-wide col tiles
    {
        const int   arow  = tr * 16 + r;
        const float mu    = smean[arow];
        const float rs    = srstd[arow];
        const float* xrp  = x + (size_t)(rowbase + arow) * DIM;

        v8f c0 = {}; v8f c1 = {};
        const bf16* bp0base = wdT + (size_t)(cb + r) * DIM;
        const bf16* bp1base = bp0base + (size_t)16 * DIM;

        for (int kb = 0; kb < DIM; kb += 32) {
            // A fragment: rows of normalized x (bf16), built on the fly
            Frag a;
            v4f x0 = *(const v4f*)(xrp + kb + 8 * h);
            v4f x1 = *(const v4f*)(xrp + kb + 8 * h + 4);
            v4f x2 = *(const v4f*)(xrp + kb + 16 + 8 * h);
            v4f x3 = *(const v4f*)(xrp + kb + 16 + 8 * h + 4);
            a.e[0]  = (bf16)((x0.x - mu) * rs); a.e[1]  = (bf16)((x0.y - mu) * rs);
            a.e[2]  = (bf16)((x0.z - mu) * rs); a.e[3]  = (bf16)((x0.w - mu) * rs);
            a.e[4]  = (bf16)((x1.x - mu) * rs); a.e[5]  = (bf16)((x1.y - mu) * rs);
            a.e[6]  = (bf16)((x1.z - mu) * rs); a.e[7]  = (bf16)((x1.w - mu) * rs);
            a.e[8]  = (bf16)((x2.x - mu) * rs); a.e[9]  = (bf16)((x2.y - mu) * rs);
            a.e[10] = (bf16)((x2.z - mu) * rs); a.e[11] = (bf16)((x2.w - mu) * rs);
            a.e[12] = (bf16)((x3.x - mu) * rs); a.e[13] = (bf16)((x3.y - mu) * rs);
            a.e[14] = (bf16)((x3.z - mu) * rs); a.e[15] = (bf16)((x3.w - mu) * rs);

            // B fragments: two contiguous 16B runs each, from transposed weights
            Frag b0, b1;
            const bf16* bp0 = bp0base + kb + 8 * h;
            const bf16* bp1 = bp1base + kb + 8 * h;
            b0.q[0] = *(const v4u*)(bp0);
            b0.q[1] = *(const v4u*)(bp0 + 16);
            b1.q[0] = *(const v4u*)(bp1);
            b1.q[1] = *(const v4u*)(bp1 + 16);

            c0 = __builtin_amdgcn_wmma_f32_16x16x32_bf16(false, a.v, false, b0.v,
                                                         (short)0, c0, false, false);
            c1 = __builtin_amdgcn_wmma_f32_16x16x32_bf16(false, a.v, false, b1.v,
                                                         (short)0, c1, false, false);
        }

        // bias + exact GELU -> H in LDS
        const float bd0 = bd_eff[cb + r];
        const float bd1 = bd_eff[cb + 16 + r];
        #pragma unroll
        for (int i = 0; i < 8; ++i) {
            int row = tr * 16 + i + 8 * h;
            Hs[row * HSTR + cb + r]      = (bf16)gelu_exact(c0[i] + bd0);
            Hs[row * HSTR + cb + 16 + r] = (bf16)gelu_exact(c1[i] + bd1);
        }
    }
    __syncthreads();

    // ---------------- Phase 3: GEMM2  out = x + scale*(H @ wu + bup) --------
    // Output 64x1024 -> row tile wv>>1, column half (wv&1)*512; 32 tiles/wave.
    {
        const int tr2 = wv >> 1;
        const int ch  = wv & 1;
        const bf16* hrow = Hs + (tr2 * 16 + r) * HSTR;

        // Hoisted A fragments (K = 0..31 and 32..63)
        Frag a0, a1;
        a0.q[0] = *(const v4u*)(hrow + 8 * h);
        a0.q[1] = *(const v4u*)(hrow + 16 + 8 * h);
        a1.q[0] = *(const v4u*)(hrow + 32 + 8 * h);
        a1.q[1] = *(const v4u*)(hrow + 48 + 8 * h);

        for (int j = 0; j < 32; ++j) {
            const int col = ch * 512 + j * 16 + r;
            const bf16* bp = wuT + (size_t)col * BOT + 8 * h;
            Frag b0, b1;
            b0.q[0] = *(const v4u*)(bp);
            b0.q[1] = *(const v4u*)(bp + 16);
            b1.q[0] = *(const v4u*)(bp + 32);   // K = 32..63
            b1.q[1] = *(const v4u*)(bp + 48);

            v8f acc = {};
            acc = __builtin_amdgcn_wmma_f32_16x16x32_bf16(false, a0.v, false, b0.v,
                                                          (short)0, acc, false, false);
            acc = __builtin_amdgcn_wmma_f32_16x16x32_bf16(false, a1.v, false, b1.v,
                                                          (short)0, acc, false, false);

            const float bu = bup[col];
            #pragma unroll
            for (int i = 0; i < 8; ++i) {
                const int row = rowbase + tr2 * 16 + i + 8 * h;
                const size_t idx = (size_t)row * DIM + col;
                out[idx] = x[idx] + scale * (acc[i] + bu);
            }
        }
    }
}

// ---------------------------------------------------------------------------
extern "C" void kernel_launch(void* const* d_in, const int* in_sizes, int n_in,
                              void* d_out, int out_size, void* d_ws, size_t ws_size,
                              hipStream_t stream)
{
    const float* x   = (const float*)d_in[0];
    const float* nw  = (const float*)d_in[1];
    const float* nb  = (const float*)d_in[2];
    const float* wd  = (const float*)d_in[3];
    const float* bd  = (const float*)d_in[4];
    const float* wu  = (const float*)d_in[5];
    const float* bu  = (const float*)d_in[6];
    const float* sc  = (const float*)d_in[7];
    float* out = (float*)d_out;

    // workspace layout: wdT (128KB bf16) | wuT (128KB bf16) | bd_eff (256B f32)
    bf16*  wdT    = (bf16*)d_ws;
    bf16*  wuT    = wdT + DIM * BOT;
    float* bd_eff = (float*)(wuT + DIM * BOT);

    prep_weights<<<(2 * DIM * BOT) / 256, 256, 0, stream>>>(wd, wu, nw, wdT, wuT);
    prep_bias<<<1, BOT, 0, stream>>>(wd, nb, bd, bd_eff);

    const int rows = in_sizes[0] / DIM;           // 16384
    adapter_kernel<<<rows / MT, 256, 0, stream>>>(x, wdT, wuT, bd_eff, bu, sc, out);
}
